// MixedSparseTraditionalMLP_8022998908954
// MI455X (gfx1250) — matrix operand order repaired and checked by reading, not compiled, not run on hardware
//
#include <hip/hip_runtime.h>
#include <hip/hip_bf16.h>

// ---------------------------------------------------------------------------
// QLoRA MLP forward for MI455X (gfx1250, wave32, WMMA, async global->LDS).
//   M = B*S = 4096, D = 4096, H = 11008, R = 16, NF4 block = 64
// Compute-bound (0.74 TFLOP f16 vs ~0.5GB HBM traffic): dequant NF4 -> f16
// once (weight slab L2-resident), two WMMA f16 GEMMs with a 3-stage
// async-to-LDS software pipeline and fused bias + rank-16 LoRA + relu.
// ---------------------------------------------------------------------------

typedef __attribute__((ext_vector_type(16))) _Float16 v16h;
typedef __attribute__((ext_vector_type(8)))  float    v8f;

static constexpr int MDIM = 4096;    // B*S
static constexpr int DDIM = 4096;
static constexpr int HDIM = 11008;
static constexpr int RDIM = 16;

static constexpr int BM = 128, BN = 128, BK = 32;
static constexpr int LSTR = 40;      // padded LDS row stride (halves): 80B = 5*16B, conflict-free

__constant__ float NF4_LUT[16] = {
    -1.0f, -0.6961928009986877f, -0.5250730514526367f, -0.39491748809814453f,
    -0.28444138169288635f, -0.18477343022823334f, -0.09105003625154495f, 0.0f,
    0.07958029955625534f, 0.16093020141124725f, 0.24611230194568634f,
    0.33791524171829224f, 0.44070982933044434f, 0.5626170039176941f,
    0.7229568362236023f, 1.0f};

// gfx1250: generic pointer to LDS carries the wave-relative LDS byte address
// in its low 32 bits (ISA 10.2).
__device__ __forceinline__ uint32_t lds_off(const void* p) {
    return (uint32_t)(uintptr_t)p;
}

__device__ __forceinline__ void async_load_b128(const void* gptr, uint32_t ldsaddr) {
    asm volatile("global_load_async_to_lds_b128 %0, %1, off"
                 :: "v"(ldsaddr), "v"((uint64_t)(uintptr_t)gptr)
                 : "memory");
}

// ---------------------------------------------------------------------------
// NF4 -> f16 dequant: 4 codes per thread, one absmax per int4 (i%64 in {0..60})
// ---------------------------------------------------------------------------
__global__ __launch_bounds__(256) void dequant_nf4_kernel(
    const int* __restrict__ codes, const float* __restrict__ absmax,
    _Float16* __restrict__ out)
{
    size_t i = ((size_t)blockIdx.x * 256 + threadIdx.x) * 4;
    int4 c = *(const int4*)(codes + i);
    float s = absmax[i >> 6];
    union { _Float16 h[4]; uint2 u; } p;
    p.h[0] = (_Float16)(NF4_LUT[c.x & 15] * s);
    p.h[1] = (_Float16)(NF4_LUT[c.y & 15] * s);
    p.h[2] = (_Float16)(NF4_LUT[c.z & 15] * s);
    p.h[3] = (_Float16)(NF4_LUT[c.w & 15] * s);
    *(uint2*)(out + i) = p.u;
}

// ---------------------------------------------------------------------------
// fp32 -> f16 activation pre-convert (so GEMM staging is a pure async copy)
// ---------------------------------------------------------------------------
__global__ __launch_bounds__(256) void f32_to_f16_kernel(
    const float* __restrict__ in, _Float16* __restrict__ out)
{
    size_t i = ((size_t)blockIdx.x * 256 + threadIdx.x) * 8;
    float4 a = *(const float4*)(in + i);
    float4 b = *(const float4*)(in + i + 4);
    union { _Float16 h[8]; uint4 u; } p;
    p.h[0] = (_Float16)a.x; p.h[1] = (_Float16)a.y;
    p.h[2] = (_Float16)a.z; p.h[3] = (_Float16)a.w;
    p.h[4] = (_Float16)b.x; p.h[5] = (_Float16)b.y;
    p.h[6] = (_Float16)b.z; p.h[7] = (_Float16)b.w;
    *(uint4*)(out + i) = p.u;
}

// ---------------------------------------------------------------------------
// Rank-16 LoRA "A" projection: u[m,r] = sum_k x[m,k] * A[k,r].
// One thread per (m,r); lanes 0..15 share m -> A rows read coalesced.
// ---------------------------------------------------------------------------
template <typename T>
__global__ __launch_bounds__(256) void lora_a_kernel(
    const T* __restrict__ x, const float* __restrict__ A,
    float* __restrict__ u, int K)
{
    int gid = blockIdx.x * 256 + threadIdx.x;
    int m = gid >> 4, r = gid & 15;
    const T* xr = x + (size_t)m * K;
    float acc = 0.f;
    for (int k = 0; k < K; k += 4) {
        acc = fmaf((float)xr[k + 0], A[(k + 0) * RDIM + r], acc);
        acc = fmaf((float)xr[k + 1], A[(k + 1) * RDIM + r], acc);
        acc = fmaf((float)xr[k + 2], A[(k + 2) * RDIM + r], acc);
        acc = fmaf((float)xr[k + 3], A[(k + 3) * RDIM + r], acc);
    }
    u[(size_t)m * RDIM + r] = acc;
}

// ---------------------------------------------------------------------------
// Tiled WMMA GEMM:  out[M,N] = act( A[M,K] * Bw[N,K]^T + bias + U[M,16]*LB[16,N] )
// Block: 256 threads = 8 waves in 4(M) x 2(N); wave tile 32x64 = 2x4 WMMAs.
// 3-stage pipeline on GLOBAL_LOAD_ASYNC_TO_LDS_B128 (4 async instr/wave/tile):
//   steady state: stage tile ks+2, s_wait_asynccnt 8 (two newest tiles may be
//   outstanding -> tile ks landed), barrier, WMMA on buffer ks%3, barrier.
// Drain iterations are peeled so the hot loop is branchless.
// Fragment loads follow ISA 7.12.2 16-bit A (K 0-7/16-23 | 8-15/24-31 split)
// and B (same K split as A) lane layouts.
// ---------------------------------------------------------------------------
template <bool RELU_H_OUT>
__global__ __launch_bounds__(256) void qlora_wmma_gemm_kernel(
    const _Float16* __restrict__ Ah,   // [M,K] f16 activations
    const _Float16* __restrict__ Bw,   // [N,K] f16 dequantized weight
    const float* __restrict__ bias,    // [N]
    const float* __restrict__ U,       // [M,16]  (x @ lora_a)
    const float* __restrict__ LB,      // [16,N]  (lora_b)
    _Float16* __restrict__ Oh, float* __restrict__ Of,
    int K, int N)
{
    __shared__ _Float16 As[3][BM * LSTR];
    __shared__ _Float16 Bs[3][BN * LSTR];

    const int tid  = threadIdx.x;
    const int lane = tid & 31;
    const int wid  = tid >> 5;
    const int l16  = lane & 15;
    const int hi   = lane >> 4;           // lane half-select
    const int wm   = (wid & 3) * 32;      // wave M offset in block tile
    const int wn   = (wid >> 2) * 64;     // wave N offset in block tile
    const int m0   = blockIdx.y * BM;
    const int n0   = blockIdx.x * BN;

    const v8f vzero = {0.f, 0.f, 0.f, 0.f, 0.f, 0.f, 0.f, 0.f};
    v8f c[2][4];
#pragma unroll
    for (int mi = 0; mi < 2; ++mi)
#pragma unroll
        for (int ni = 0; ni < 4; ++ni) c[mi][ni] = vzero;

    // stage one BMxBK + BNxBK tile pair: 4 async b128 instructions per wave
    auto stage_tile = [&](int ksx, int buf) {
#pragma unroll
        for (int i = 0; i < 2; ++i) {              // A: 512 chunks of 8 halves
            int ch = tid + i * 256;
            int row = ch >> 2, col = (ch & 3) << 3;
            async_load_b128(Ah + (size_t)(m0 + row) * K + ksx * BK + col,
                            lds_off(&As[buf][row * LSTR + col]));
        }
#pragma unroll
        for (int i = 0; i < 2; ++i) {              // B: 512 chunks of 8 halves
            int ch = tid + i * 256;
            int row = ch >> 2, col = (ch & 3) << 3;
            async_load_b128(Bw + (size_t)(n0 + row) * K + ksx * BK + col,
                            lds_off(&Bs[buf][row * LSTR + col]));
        }
    };

    // WMMA on one landed LDS buffer (caller has barriered)
    auto compute_tile = [&](int buf) {
        union Frag { v16h v; uint4 q[2]; };
        Frag a[2], b[4];
#pragma unroll
        for (int mi = 0; mi < 2; ++mi) {
            const int row = wm + mi * 16 + l16;
            // lanes 0-15: K=0..7 & 16..23 ; lanes 16-31: K=8..15 & 24..31
            a[mi].q[0] = *(const uint4*)&As[buf][row * LSTR + hi * 8];
            a[mi].q[1] = *(const uint4*)&As[buf][row * LSTR + hi * 8 + 16];
        }
#pragma unroll
        for (int ni = 0; ni < 4; ++ni) {
            const int row = wn + ni * 16 + l16;
            // lanes 0-15: K=0..7 & 16..23 ; lanes 16-31: K=8..15 & 24..31
            b[ni].q[0] = *(const uint4*)&Bs[buf][row * LSTR + hi * 8];
            b[ni].q[1] = *(const uint4*)&Bs[buf][row * LSTR + hi * 8 + 16];
        }
#pragma unroll
        for (int mi = 0; mi < 2; ++mi)
#pragma unroll
            for (int ni = 0; ni < 4; ++ni)
                c[mi][ni] = __builtin_amdgcn_wmma_f32_16x16x32_f16(
                    false, a[mi].v, false, b[ni].v,
                    (short)0, c[mi][ni], false, false);
    };

    const int kSteps = K / BK;                     // >= 2 for all our shapes

    // ---- prologue: two tiles in flight ----
    stage_tile(0, 0);
    stage_tile(1, 1);

    // ---- branchless steady state ----
    int bufc = 0, bufs = 2;                        // compute buf = ks%3, stage buf = (ks+2)%3
    for (int ks = 0; ks < kSteps - 2; ++ks) {
        stage_tile(ks + 2, bufs);
        asm volatile("s_wait_asynccnt 0x8" ::: "memory");   // tile ks landed
        __syncthreads();
        compute_tile(bufc);
        __syncthreads();                                    // protect buffer reuse
        bufc = (bufc == 2) ? 0 : bufc + 1;
        bufs = (bufs == 2) ? 0 : bufs + 1;
    }

    // ---- peeled drain: tiles kSteps-2 and kSteps-1 ----
    asm volatile("s_wait_asynccnt 0x4" ::: "memory");
    __syncthreads();
    compute_tile(bufc);
    __syncthreads();
    bufc = (bufc == 2) ? 0 : bufc + 1;
    asm volatile("s_wait_asynccnt 0x0" ::: "memory");
    __syncthreads();
    compute_tile(bufc);

    // ---- epilogue: bias + rank-16 LoRA (+relu) ----
    // C/D layout: VGPR vr, lanes 0-15 -> M=vr, N=l16 ; lanes 16-31 -> M=vr+8.
#pragma unroll
    for (int ni = 0; ni < 4; ++ni) {
        const int n = n0 + wn + ni * 16 + l16;
        const float bn = bias[n];
        float lbc[RDIM];
#pragma unroll
        for (int r = 0; r < RDIM; ++r) lbc[r] = LB[(size_t)r * N + n];
#pragma unroll
        for (int mi = 0; mi < 2; ++mi) {
#pragma unroll
            for (int vr = 0; vr < 8; ++vr) {
                const int m = m0 + wm + mi * 16 + hi * 8 + vr;
                const float* ur = U + (size_t)m * RDIM;
                float acc = c[mi][ni][vr] + bn;
#pragma unroll
                for (int r = 0; r < RDIM; ++r) acc = fmaf(ur[r], lbc[r], acc);
                if constexpr (RELU_H_OUT) {
                    acc = acc > 0.f ? acc : 0.f;
                    Oh[(size_t)m * N + n] = (_Float16)acc;
                } else {
                    Of[(size_t)m * N + n] = acc;
                }
            }
        }
    }
}

// ---------------------------------------------------------------------------
// Launch: 7 stream-ordered kernels. Workspace layout (all 256B aligned):
//   [0,      90.2MB)   f16 weight slab (reused for up then down)
//   [90.2,  180.4MB)   x2 (relu output) as f16, [M,H]
//   [180.4, 213.9MB)   x1 as f16, [M,D]
//   [213.9, +256KB)    u = rank-16 intermediate, [M,16] f32 (reused)
// ---------------------------------------------------------------------------
extern "C" void kernel_launch(void* const* d_in, const int* in_sizes, int n_in,
                              void* d_out, int out_size, void* d_ws, size_t ws_size,
                              hipStream_t stream)
{
    (void)in_sizes; (void)n_in; (void)out_size; (void)ws_size;

    const float* x1  = (const float*)d_in[0];
    const int*   wuc = (const int*)  d_in[1];
    const float* wua = (const float*)d_in[2];
    const float* bup = (const float*)d_in[3];
    const float* lua = (const float*)d_in[4];   // [D,16]
    const float* lub = (const float*)d_in[5];   // [16,H]
    const int*   wdc = (const int*)  d_in[6];
    const float* wda = (const float*)d_in[7];
    const float* bdn = (const float*)d_in[8];
    const float* lda = (const float*)d_in[9];   // [H,16]
    const float* ldb = (const float*)d_in[10];  // [16,D]
    float* out = (float*)d_out;

    const size_t NW  = (size_t)HDIM * DDIM;     // 45,088,768 (multiple of 1024)
    const size_t MH  = (size_t)MDIM * HDIM;
    const size_t MD  = (size_t)MDIM * DDIM;
    char* ws = (char*)d_ws;
    _Float16* wf16 = (_Float16*)ws;                          // NW halves
    _Float16* x2   = (_Float16*)(ws + NW * 2);               // MH halves
    _Float16* x1h  = (_Float16*)(ws + NW * 2 + MH * 2);      // MD halves
    float*    u    = (float*)   (ws + NW * 2 + MH * 2 + MD * 2);

    const int dequant_blocks = (int)(NW / 1024);             // 4 elems/thread
    const int conv_blocks    = (int)(MD / 2048);             // 8 elems/thread
    const int lora_blocks    = (MDIM * RDIM) / 256;

    // ---- up path ----
    f32_to_f16_kernel<<<conv_blocks, 256, 0, stream>>>(x1, x1h);
    dequant_nf4_kernel<<<dequant_blocks, 256, 0, stream>>>(wuc, wua, wf16);
    lora_a_kernel<float><<<lora_blocks, 256, 0, stream>>>(x1, lua, u, DDIM);
    qlora_wmma_gemm_kernel<true>
        <<<dim3(HDIM / BN, MDIM / BM), 256, 0, stream>>>(
            x1h, wf16, bup, u, lub, x2, nullptr, DDIM, HDIM);

    // ---- down path (weight slab + u reused; stream order guarantees safety) ----
    dequant_nf4_kernel<<<dequant_blocks, 256, 0, stream>>>(wdc, wda, wf16);
    lora_a_kernel<_Float16><<<lora_blocks, 256, 0, stream>>>(x2, lda, u, HDIM);
    qlora_wmma_gemm_kernel<false>
        <<<dim3(DDIM / BN, MDIM / BM), 256, 0, stream>>>(
            x2, wf16, bdn, u, ldb, nullptr, out, HDIM, DDIM);
}
